// SelfAttention_48893907698025
// MI455X (gfx1250) — compile-verified, matrix operand order checked by reading, not run on hardware
//
#include <hip/hip_runtime.h>

typedef _Float16 v16h __attribute__((ext_vector_type(16)));
typedef _Float16 v8h  __attribute__((ext_vector_type(8)));
typedef float    v8f  __attribute__((ext_vector_type(8)));
typedef float    v4f  __attribute__((ext_vector_type(4)));

#define BATCH 4
#define CDIM  256
#define NSP   4096
#define DK    32
#define CTILES 4   // 16-wide channel tiles per wave in flash_attn (64 channels/wave)

// Load a 16x16x32-f16 WMMA A/B operand fragment.
// Per-lane half->K mapping (ISA 7.12.2, 16-bit A 16x32):
//   lanes 0-15 : halves 0-7 = k 0..7,  halves 8-15 = k 16..23
//   lanes 16-31: halves 0-7 = k 8..15, halves 8-15 = k 24..31
// `row` points at a contiguous 32-halfword K-major row; lhalf = lane>=16.
static __device__ __forceinline__ v16h load_ab16(const _Float16* __restrict__ row, int lhalf) {
    v8h lo = *(const v8h*)(row + lhalf * 8);
    v8h hi = *(const v8h*)(row + 16 + lhalf * 8);
    v16h r;
#pragma unroll
    for (int t = 0; t < 8; ++t) { r[t] = lo[t]; r[8 + t] = hi[t]; }
    return r;
}

__global__ void cvt_f32_f16(const float* __restrict__ src, _Float16* __restrict__ dst, int n) {
    int i = blockIdx.x * blockDim.x + threadIdx.x;
    if (i < n) dst[i] = (_Float16)src[i];
}

// x [B][C][N] f32  ->  Xh [B][N][C] f16
__global__ void transpose_x(const float* __restrict__ x, _Float16* __restrict__ Xh) {
    int i = blockIdx.x * blockDim.x + threadIdx.x;
    if (i >= BATCH * CDIM * NSP) return;
    int sp = i % NSP;
    int c  = (i / NSP) % CDIM;
    int b  = i / (NSP * CDIM);
    Xh[((size_t)b * NSP + sp) * CDIM + c] = (_Float16)x[i];
}

// D[m, n] = sum_c Xh[b][m][c] * Wt[n][c] + bias[n]
// mode 0: Out[b][m][nout] (f16, row-major over nout)      -> used for Q, K
// mode 1: Out[b][n][NSP]  (f16, n-major; packed b128 st)  -> used for V (= h natural layout)
__global__ __launch_bounds__(256) void proj_gemm(
    const _Float16* __restrict__ Xh, const _Float16* __restrict__ Wt,
    const float* __restrict__ bias, _Float16* __restrict__ Out,
    int nout, int mode)
{
    const int wave  = threadIdx.x >> 5;
    const int lane  = threadIdx.x & 31;
    const int l16   = lane & 15;
    const int lhalf = lane >> 4;
    const int b     = blockIdx.y;
    const int tilesN = nout >> 4;
    const int t      = blockIdx.x * 8 + wave;
    const int mbase  = (t / tilesN) * 16;
    const int nbase  = (t % tilesN) * 16;

    const _Float16* arow = Xh + ((size_t)b * NSP + mbase + l16) * CDIM;
    const _Float16* brow = Wt + (size_t)(nbase + l16) * CDIM;

    v8f acc = {};
#pragma unroll
    for (int cb = 0; cb < CDIM; cb += 32) {
        v16h a  = load_ab16(arow + cb, lhalf);
        v16h bb = load_ab16(brow + cb, lhalf);
        acc = __builtin_amdgcn_wmma_f32_16x16x32_f16(false, a, false, bb, (short)0, acc, false, false);
    }
    const float bv = bias[nbase + l16];
    if (mode == 0) {
        // element r: m = mbase + r + 8*lhalf, n = nbase + l16
        _Float16* o = Out + ((size_t)b * NSP + mbase + 8 * lhalf) * nout + (nbase + l16);
#pragma unroll
        for (int r = 0; r < 8; ++r) o[(size_t)r * nout] = (_Float16)(acc[r] + bv);
    } else {
        v8h pk;
#pragma unroll
        for (int r = 0; r < 8; ++r) pk[r] = (_Float16)(acc[r] + bv);
        _Float16* o = Out + ((size_t)b * nout + nbase + l16) * NSP + mbase + 8 * lhalf;
        *(v8h*)o = pk;
    }
}

// Flash attention + residual epilogue.
// Block = 4 waves; block handles 16 queries x all 256 channels (wave w -> channels [64w, 64w+64)).
// S is computed TRANSPOSED (A = K tile, B = Q) so the probability fragment lands
// directly in the A-operand layout for the P*V WMMA (no LDS transpose needed).
__global__ __launch_bounds__(128) void flash_attn(
    const _Float16* __restrict__ Qh, const _Float16* __restrict__ Kh,
    const _Float16* __restrict__ Vh, const float* __restrict__ x,
    const float* __restrict__ gamma, float* __restrict__ out)
{
    const int wave  = threadIdx.x >> 5;
    const int lane  = threadIdx.x & 31;
    const int l16   = lane & 15;
    const int lhalf = lane >> 4;
    const int b     = blockIdx.y;
    const int ibase = blockIdx.x * 16;
    const int c0    = wave * (CTILES * 16);

    // Q B-operand: lane = query column, loaded once, reused for every key chunk
    const v16h qb = load_ab16(Qh + ((size_t)b * NSP + ibase + l16) * DK, lhalf);

    const _Float16* krows = Kh + (size_t)b * NSP * DK;
    const _Float16* vrow[CTILES];
#pragma unroll
    for (int t = 0; t < CTILES; ++t)
        vrow[t] = Vh + ((size_t)b * CDIM + c0 + t * 16 + l16) * NSP;

    v8f o[CTILES];
#pragma unroll
    for (int t = 0; t < CTILES; ++t) o[t] = (v8f){};
    float mprev = -3.0e38f;   // running row max (query = lane%16)
    float lsum  = 0.0f;       // running row sum

    for (int jb = 0; jb < NSP; jb += 32) {
        // Stage this chunk's operand loads up front so VMEM overlaps the softmax VALU.
        v16h ka0 = load_ab16(krows + (size_t)(jb + l16) * DK, lhalf);
        v16h ka1 = load_ab16(krows + (size_t)(jb + 16 + l16) * DK, lhalf);
        v16h vb[CTILES];
#pragma unroll
        for (int t = 0; t < CTILES; ++t) vb[t] = load_ab16(vrow[t] + jb, lhalf);

        // Speculative prefetch of next chunk's K rows (global_prefetch_b8).
        __builtin_prefetch(krows + (size_t)(jb + 32 + l16) * DK, 0, 1);

        // S^T tiles: St[key, query]; two 16(key)x16(query) WMMAs, k-dim = d = 32
        v8f z = {};
        v8f st0 = __builtin_amdgcn_wmma_f32_16x16x32_f16(false, ka0, false, qb, (short)0, z, false, false);
        v8f st1 = __builtin_amdgcn_wmma_f32_16x16x32_f16(false, ka1, false, qb, (short)0, z, false, false);

        // Online softmax per query (= lane%16); lanes q and q+16 hold complementary key halves.
        float lm = st0[0];
#pragma unroll
        for (int r = 0; r < 8; ++r) { lm = fmaxf(lm, st0[r]); lm = fmaxf(lm, st1[r]); }
        lm = fmaxf(lm, __shfl_xor(lm, 16, 32));
        const float mnew = fmaxf(mprev, lm);
        const float sc   = __expf(mprev - mnew);

        float rs = 0.0f;
        v16h pa;  // probabilities, directly in A-operand layout for P*V
#pragma unroll
        for (int r = 0; r < 8; ++r) {
            float p0 = __expf(st0[r] - mnew);
            float p1 = __expf(st1[r] - mnew);
            rs += p0 + p1;
            pa[r]     = (_Float16)p0;
            pa[8 + r] = (_Float16)p1;
        }
        rs += __shfl_xor(rs, 16, 32);
        lsum = lsum * sc + rs;
        mprev = mnew;

        // Rescale accumulators: O element r belongs to query (r + 8*lhalf)
#pragma unroll
        for (int r = 0; r < 8; ++r) {
            float s = __shfl(sc, r + 8 * lhalf, 32);
#pragma unroll
            for (int t = 0; t < CTILES; ++t) o[t][r] *= s;
        }

#pragma unroll
        for (int t = 0; t < CTILES; ++t)
            o[t] = __builtin_amdgcn_wmma_f32_16x16x32_f16(false, pa, false, vb[t], (short)0, o[t], false, false);
    }

    const float g    = gamma[0];
    const float linv = 1.0f / lsum;
    v8f il;
#pragma unroll
    for (int r = 0; r < 8; ++r) il[r] = __shfl(linv, r + 8 * lhalf, 32);

    // out[b][c][i] = x + gamma * O[i,c]/l ; element r: i = ibase + 8*lhalf + r, c fixed per lane
#pragma unroll
    for (int t = 0; t < CTILES; ++t) {
        size_t idx = ((size_t)b * CDIM + c0 + t * 16 + l16) * NSP + ibase + 8 * lhalf;
        v4f xa = *(const v4f*)(x + idx);
        v4f xb = *(const v4f*)(x + idx + 4);
        v4f ra, rb;
#pragma unroll
        for (int r = 0; r < 4; ++r) {
            ra[r] = xa[r] + g * o[t][r] * il[r];
            rb[r] = xb[r] + g * o[t][4 + r] * il[4 + r];
        }
        *(v4f*)(out + idx)     = ra;
        *(v4f*)(out + idx + 4) = rb;
    }
}

extern "C" void kernel_launch(void* const* d_in, const int* in_sizes, int n_in,
                              void* d_out, int out_size, void* d_ws, size_t ws_size,
                              hipStream_t stream) {
    (void)in_sizes; (void)n_in; (void)out_size; (void)ws_size;
    const float* x     = (const float*)d_in[0];
    const float* Wf    = (const float*)d_in[1];
    const float* bf    = (const float*)d_in[2];
    const float* Wg    = (const float*)d_in[3];
    const float* bg    = (const float*)d_in[4];
    const float* Wh    = (const float*)d_in[5];
    const float* bh    = (const float*)d_in[6];
    const float* gamma = (const float*)d_in[7];
    float* out = (float*)d_out;

    char* ws = (char*)d_ws;
    _Float16* Xh  = (_Float16*)(ws);                                  // 8 MB  [B][N][C]
    _Float16* Qh  = (_Float16*)(ws + (size_t)8  * 1024 * 1024);       // 1 MB  [B][N][32]
    _Float16* Kh  = (_Float16*)(ws + (size_t)9  * 1024 * 1024);       // 1 MB  [B][N][32]
    _Float16* Vh  = (_Float16*)(ws + (size_t)10 * 1024 * 1024);       // 8 MB  [B][C][N]
    _Float16* Wfh = (_Float16*)(ws + (size_t)18 * 1024 * 1024);                 // 16 KB
    _Float16* Wgh = (_Float16*)(ws + (size_t)18 * 1024 * 1024 + 32 * 1024);     // 16 KB
    _Float16* Whh = (_Float16*)(ws + (size_t)18 * 1024 * 1024 + 64 * 1024);     // 128 KB

    // 1) weight conversion to f16
    cvt_f32_f16<<<dim3((DK * CDIM + 255) / 256), dim3(256), 0, stream>>>(Wf, Wfh, DK * CDIM);
    cvt_f32_f16<<<dim3((DK * CDIM + 255) / 256), dim3(256), 0, stream>>>(Wg, Wgh, DK * CDIM);
    cvt_f32_f16<<<dim3((CDIM * CDIM + 255) / 256), dim3(256), 0, stream>>>(Wh, Whh, CDIM * CDIM);

    // 2) x -> Xh (transpose + f16)
    transpose_x<<<dim3((BATCH * CDIM * NSP) / 256), dim3(256), 0, stream>>>(x, Xh);

    // 3) projections (WMMA GEMMs). tiles = (N/16)*(nout/16), 8 tiles per block.
    proj_gemm<<<dim3((NSP / 16) * (DK / 16) / 8, BATCH), dim3(256), 0, stream>>>(Xh, Wfh, bf, Qh, DK, 0);
    proj_gemm<<<dim3((NSP / 16) * (DK / 16) / 8, BATCH), dim3(256), 0, stream>>>(Xh, Wgh, bg, Kh, DK, 0);
    proj_gemm<<<dim3((NSP / 16) * (CDIM / 16) / 8, BATCH), dim3(256), 0, stream>>>(Xh, Whh, bh, Vh, CDIM, 1);

    // 4) flash attention + residual (4 waves/block, 64 channels per wave)
    flash_attn<<<dim3(NSP / 16, BATCH), dim3(128), 0, stream>>>(Qh, Kh, Vh, x, gamma, out);
}
